// TransformerEncoderLayer_11544872091750
// MI455X (gfx1250) — matrix operand order repaired and checked by reading, not compiled, not run on hardware
//
#include <hip/hip_runtime.h>

#define S_LEN 1024
#define BSZ   4
#define NB    8
#define BD    128
#define NH    4
#define HD    32
#define EMB   1024
#define FBD   512

typedef __attribute__((ext_vector_type(16))) __bf16 v16bf;
typedef __attribute__((ext_vector_type(2)))  __bf16 v2bf;
typedef __attribute__((ext_vector_type(8)))  float  v8f;
typedef __attribute__((ext_vector_type(2)))  float  v2f;

union Frag {
  v16bf          v;
  unsigned short u[16];
  unsigned int   w[8];
  uint4          x4[2];
};

static __device__ __forceinline__ unsigned short f2bf(float f) {
  __bf16 h = (__bf16)f;                      // native v_cvt -> bf16
  return __builtin_bit_cast(unsigned short, h);
}
static __device__ __forceinline__ unsigned pk2bf(float a, float b) {
  v2f t = {a, b};                            // single v_cvt_pk_bf16_f32
  return __builtin_bit_cast(unsigned, __builtin_convertvector(t, v2bf));
}
static __device__ __forceinline__ float bf2f(unsigned short h) {
  return __builtin_bit_cast(float, ((unsigned int)h) << 16);
}
static __device__ __forceinline__ v8f wmma_bf16(v16bf a, v16bf b, v8f c) {
  return __builtin_amdgcn_wmma_f32_16x16x32_bf16(false, a, false, b, (short)0, c,
                                                 false, false);
}

// ---------------------------------------------------------------------------
// Weight pack: f32 (G, Kd, N) -> bf16 transposed, out[g*gStride + (jOff+j)*Kd + k]
// ---------------------------------------------------------------------------
__global__ void pack_w(const float* __restrict__ W, unsigned short* __restrict__ Wt,
                       int Kd, int N, int gStride, int jOff) {
  int idx = blockIdx.x * 256 + threadIdx.x;      // over G*N*Kd inputs
  int k = idx % Kd;
  int j = (idx / Kd) % N;
  int g = idx / (Kd * N);
  Wt[g * gStride + (jOff + j) * Kd + k] = f2bf(W[(g * Kd + k) * N + j]);
}

__global__ void pack_bias(const float* __restrict__ src, float* __restrict__ dst,
                          int N, int gStride, int off, int total) {
  int idx = blockIdx.x * 256 + threadIdx.x;
  if (idx >= total) return;
  int g = idx / N, j = idx - g * N;
  dst[g * gStride + off + j] = src[idx];
}

// ---------------------------------------------------------------------------
// Grouped GEMM:  C[m, n, j] = A[m, n, :] @ W[n, :, j] + bias[n, j]
//   A:  (M, G, KD)  f32 or bf16 (AT = float | unsigned short)
//   Wt: (G, N, KD)  bf16 (pre-transposed) -> B-frag = two b128 loads
//   EP: 0 none, 1 relu; KD compile-time -> fully unrolled K loop
//   OM: 0 f32 out (M,G,N), 1 bf16 out (M,G,N),
//       3 fused QKV out: j<128 -> q*scale, <256 -> k, else V-transposed
// grid = (M/16, G), block = 256 (8 waves); A tile staged bf16 in LDS
// (bf16 A uses global_load_async_to_lds_b128 DMA copies).
// ---------------------------------------------------------------------------
template <int EP, int OM, int KD, typename AT>
__global__ __launch_bounds__(256) void gemm_grouped(
    const AT* __restrict__ A, const unsigned short* __restrict__ Wt,
    const float* __restrict__ bias, float* __restrict__ Cf,
    unsigned short* __restrict__ Cb, int N, int G, float scale)
{
  __shared__ unsigned short sA[16 * KD];
  const int tid = threadIdx.x;
  const int m0  = blockIdx.x * 16;
  const int n   = blockIdx.y;

  constexpr int tot = 16 * KD;                    // multiple of 1024
  if constexpr (sizeof(AT) == 2) {                // bf16 A: async DMA to LDS
    for (int i = tid * 8; i < tot; i += 2048) {
      int row = i / KD, k = i - row * KD;
      const AT* src = A + ((m0 + row) * G + n) * KD + k;
      unsigned loff = (unsigned)(unsigned long long)(void*)(sA + i);
      asm volatile("global_load_async_to_lds_b128 %0, %1, off"
                   :: "v"(loff), "v"(src) : "memory");
    }
    asm volatile("s_wait_asynccnt 0" ::: "memory");
  } else {                                        // f32 A: load4 + pack bf16
    for (int i = tid * 4; i < tot; i += 1024) {
      int row = i / KD, k = i - row * KD;
      const float4 va = *(const float4*)(A + ((m0 + row) * G + n) * KD + k);
      uint2 pk;
      pk.x = pk2bf(va.x, va.y);
      pk.y = pk2bf(va.z, va.w);
      *(uint2*)(sA + i) = pk;
    }
  }
  __syncthreads();

  const int wave = tid >> 5, lane = tid & 31;
  const int half = lane >> 4, col = lane & 15;
  const unsigned int* sA32 = (const unsigned int*)sA;
  const size_t PEL = (size_t)S_LEN * BSZ * EMB;   // element stride between q/k/v

  for (int j0 = wave * 16; j0 < N; j0 += 128) {
    v8f acc = {0.f, 0.f, 0.f, 0.f, 0.f, 0.f, 0.f, 0.f};
    const unsigned short* wrow = Wt + (n * N + j0 + col) * KD;
#pragma unroll
    for (int kk = 0; kk < KD; kk += 32) {
      Frag fa;                                    // A-frag: two ds_read_b128
      int abase = ((col * KD + kk) >> 1) + half * 4;
      fa.x4[0] = *(const uint4*)(sA32 + abase);
      fa.x4[1] = *(const uint4*)(sA32 + abase + 8);
      Frag fb;                                    // B-frag: two b128 loads
      const uint4* wp = (const uint4*)(wrow + kk);
      fb.x4[0] = wp[half];
      fb.x4[1] = wp[half + 2];
      acc = wmma_bf16(fa.v, fb.v, acc);
    }
    float bb = bias[n * N + j0 + col];
#pragma unroll
    for (int r = 0; r < 8; ++r) {
      float vv = acc[r] + bb;
      if constexpr (EP == 1) vv = fmaxf(vv, 0.0f);
      int m = m0 + r + half * 8;
      if constexpr (OM == 0) {
        Cf[(m * G + n) * N + j0 + col] = vv;
      } else if constexpr (OM == 1) {
        Cb[(m * G + n) * N + j0 + col] = f2bf(vv);
      } else {                                    // OM == 3: fused QKV
        int j = j0 + col;
        int sec = j >> 7, jl = j & 127;
        if (sec == 0) {
          Cb[(m * G + n) * BD + jl] = f2bf(vv * scale);
        } else if (sec == 1) {
          Cb[PEL + (m * G + n) * BD + jl] = f2bf(vv);
        } else {                                  // V-transpose: (b,n,h,d,s)
          int s = m >> 2, b = m & 3;
          int h = jl >> 5, d = jl & 31;
          Cb[2 * PEL + (((b * NB + n) * NH + h) * HD + d) * S_LEN + s] = f2bf(vv);
        }
      }
    }
  }
}

// ---------------------------------------------------------------------------
// Block-diagonal attention, two-pass WMMA, gate fused into the epilogue.
// qb/kb: (S,B,NB,BD) bf16;  vt: (B,NB,NH,HD,S) bf16; outb: (S,B,NB,BD) bf16.
// Scores kept bf16 in LDS as [key][row] so pass-1 spills are ds_store_b128.
// grid = (S/16, NB*NH, B), block = 32.
// ---------------------------------------------------------------------------
__global__ __launch_bounds__(32) void attn_fa(
    const unsigned short* __restrict__ qb, const unsigned short* __restrict__ kb,
    const unsigned short* __restrict__ vt, const unsigned char* __restrict__ pmask,
    const float* __restrict__ gate, unsigned short* __restrict__ outb)
{
  __shared__ unsigned short sS[S_LEN * 16];      // [key][row] bf16, 32 KB
  __shared__ float sm[16], sl[16];
  const int qt = blockIdx.x, nh = blockIdx.y, b = blockIdx.z;
  const int n = nh >> 2, h = nh & 3;
  const int lane = threadIdx.x, half = lane >> 4, col = lane & 15;

  Frag fq;                                       // Q A-frag, row = col
  {
    int base = (((qt * 16 + col) * BSZ + b) * NB + n) * BD + h * HD;
    const uint4* qp = (const uint4*)(qb + base);
    fq.x4[0] = qp[half];                         // d = half*8 .. +7
    fq.x4[1] = qp[2 + half];                     // d = 16+half*8 .. +7
  }

  float mloc[8];
#pragma unroll
  for (int r = 0; r < 8; ++r) mloc[r] = -3.0e38f;

  // pass 1: scores = Q K^T (masked) -> bf16 LDS, running row max
  for (int kt = 0; kt < S_LEN / 16; ++kt) {
    int key = kt * 16 + col;
    Frag fk;
    {
      int kbase = ((key * BSZ + b) * NB + n) * BD + h * HD;
      const uint4* kp = (const uint4*)(kb + kbase);
      fk.x4[0] = kp[half];                       // d = half*8 .. +7
      fk.x4[1] = kp[2 + half];                   // d = 16+half*8 .. +7
    }
    v8f sc = {0.f, 0.f, 0.f, 0.f, 0.f, 0.f, 0.f, 0.f};
    sc = wmma_bf16(fq.v, fk.v, sc);
    bool mk = pmask[b * S_LEN + key] != 0;
    float fs[8];
#pragma unroll
    for (int r = 0; r < 8; ++r) {
      fs[r] = mk ? -1.0e8f : sc[r];
      mloc[r] = fmaxf(mloc[r], fs[r]);
    }
    Frag st;                                     // rows half*8..+7 contiguous
#pragma unroll
    for (int p = 0; p < 4; ++p)
      st.w[p] = pk2bf(fs[2 * p], fs[2 * p + 1]);
    *(uint4*)(sS + key * 16 + half * 8) = st.x4[0];
  }
#pragma unroll
  for (int r = 0; r < 8; ++r)
    for (int msk = 1; msk < 16; msk <<= 1)
      mloc[r] = fmaxf(mloc[r], __shfl_xor(mloc[r], msk));
  if (col == 0) {
#pragma unroll
    for (int r = 0; r < 8; ++r) sm[r + half * 8] = mloc[r];
  }
  __syncthreads();

  // pass 2: P = exp(S - max), O = P V
  float rowmax = sm[col];
  float lsum = 0.0f;
  v8f o0 = {0.f, 0.f, 0.f, 0.f, 0.f, 0.f, 0.f, 0.f};
  v8f o1 = {0.f, 0.f, 0.f, 0.f, 0.f, 0.f, 0.f, 0.f};
  const int hb = ((b * NB + n) * NH + h) * HD;
  for (int kc = 0; kc < S_LEN / 32; ++kc) {
    Frag fp;
#pragma unroll
    for (int p = 0; p < 8; ++p) {                // e = 2p, 2p+1 -> adjacent keys
      int e0 = 2 * p;
      int key0 = kc * 32 + e0 + half * 16;
      float p0 = __expf(bf2f(sS[key0 * 16 + col]) - rowmax);
      float p1 = __expf(bf2f(sS[(key0 + 1) * 16 + col]) - rowmax);
      lsum += p0 + p1;
      fp.w[p] = pk2bf(p0, p1);
    }
    Frag fv0, fv1;
    {
      const uint4* vp0 = (const uint4*)(vt + (hb + col) * S_LEN + kc * 32 + half * 16);
      const uint4* vp1 = (const uint4*)(vt + (hb + 16 + col) * S_LEN + kc * 32 + half * 16);
      fv0.x4[0] = vp0[0]; fv0.x4[1] = vp0[1];
      fv1.x4[0] = vp1[0]; fv1.x4[1] = vp1[1];
    }
    o0 = wmma_bf16(fp.v, fv0.v, o0);
    o1 = wmma_bf16(fp.v, fv1.v, o1);
  }
  lsum += __shfl_xor(lsum, 16);
  if (half == 0) sl[col] = lsum;
  __syncthreads();
#pragma unroll
  for (int r = 0; r < 8; ++r) {
    int srow = qt * 16 + r + half * 8;
    float gv  = gate[(srow * BSZ + b) * (NB * NH) + n * NH + h];
    float inv = gv / sl[r + half * 8];
    int base  = ((srow * BSZ + b) * NB + n) * BD + h * HD;
    outb[base + col]      = f2bf(o0[r] * inv);
    outb[base + 16 + col] = f2bf(o1[r] * inv);
  }
}

// ---------------------------------------------------------------------------
// Gate: logits = x_b @ Wg, softmax over 32 (n,h) slots, keep top-16.
// ---------------------------------------------------------------------------
__global__ __launch_bounds__(256) void gate_topk(
    const float* __restrict__ x, const float* __restrict__ Wg,
    float* __restrict__ gate)
{
  int pos  = blockIdx.x * 8 + (threadIdx.x >> 5);
  int lane = threadIdx.x & 31;
  int n = lane >> 2, h = lane & 3;
  const float* xp = x + (pos * NB + n) * BD;
  const float* wp = Wg + n * BD * NH + h;
  float acc = 0.0f;
  for (int d = 0; d < BD; ++d) acc += xp[d] * wp[d * NH];
  float mx = acc;
  for (int msk = 1; msk < 32; msk <<= 1) mx = fmaxf(mx, __shfl_xor(mx, msk));
  float e = __expf(acc - mx);
  float sum = e;
  for (int msk = 1; msk < 32; msk <<= 1) sum += __shfl_xor(sum, msk);
  float g = e / sum;
  int ge = 0;                                   // #{j : g_j >= g_lane}
  for (int j = 0; j < 32; ++j) ge += (__shfl(g, j) >= g) ? 1 : 0;
  float cand = (ge >= 16) ? g : -1.0f;          // kth largest (k = 16)
  for (int msk = 1; msk < 32; msk <<= 1) cand = fmaxf(cand, __shfl_xor(cand, msk));
  gate[pos * 32 + lane] = (g >= cand) ? g : 0.0f;
}

// ---------------------------------------------------------------------------
// Inter-block comm attention on the fused (Mc, 192) q|k|v buffer.
// ---------------------------------------------------------------------------
__global__ __launch_bounds__(256) void comm_attn(
    const float* __restrict__ cqkv, float* __restrict__ cout)
{
  int idx = blockIdx.x * 256 + threadIdx.x;     // (pos, n, h)
  int pos = idx >> 4, n = (idx >> 1) & 7, h = idx & 1;
  const float* base = cqkv + (size_t)pos * (NB * 192);
  const float* qp = base + n * 192 + h * 32;
  float sc[8], mx = -3.0e38f;
  for (int m = 0; m < 8; ++m) {
    const float* kp = base + m * 192 + 64 + h * 32;
    float a = 0.0f;
    for (int d = 0; d < 32; ++d) a += qp[d] * kp[d];
    sc[m] = a * 0.17677669529663687f;           // COMM_HD^-0.5
    mx = fmaxf(mx, sc[m]);
  }
  float sum = 0.0f;
  for (int m = 0; m < 8; ++m) { sc[m] = __expf(sc[m] - mx); sum += sc[m]; }
  float inv = 1.0f / sum;
  float* op = cout + (pos * NB + n) * 64 + h * 32;
  for (int d = 0; d < 32; ++d) {
    float a = 0.0f;
    for (int m = 0; m < 8; ++m) a += sc[m] * base[m * 192 + 128 + h * 32 + d];
    op[d] = a * inv;
  }
}

// ---------------------------------------------------------------------------
// Fused residual add + per-block LayerNorm (128 features). Wave per row.
// ---------------------------------------------------------------------------
__global__ __launch_bounds__(256) void add_norm(
    const float* __restrict__ a, const float* __restrict__ b,
    const float* __restrict__ w, const float* __restrict__ bias,
    float* __restrict__ out)
{
  int row  = blockIdx.x * 8 + (threadIdx.x >> 5);
  int lane = threadIdx.x & 31;
  const float4* a4 = (const float4*)a;
  const float4* b4 = (const float4*)b;
  float4 va = a4[row * 32 + lane];
  float4 vb = b4[row * 32 + lane];
  float sx = va.x + vb.x, sy = va.y + vb.y, sz = va.z + vb.z, sw = va.w + vb.w;
  float sum = sx + sy + sz + sw;
  for (int msk = 1; msk < 32; msk <<= 1) sum += __shfl_xor(sum, msk);
  float mu = sum * (1.0f / 128.0f);
  float dx = sx - mu, dy = sy - mu, dz = sz - mu, dw = sw - mu;
  float vs = dx * dx + dy * dy + dz * dz + dw * dw;
  for (int msk = 1; msk < 32; msk <<= 1) vs += __shfl_xor(vs, msk);
  float rstd = rsqrtf(vs * (1.0f / 128.0f) + 1e-5f);
  int f = lane * 4;
  float4 o;
  o.x = dx * rstd * w[f + 0] + bias[f + 0];
  o.y = dy * rstd * w[f + 1] + bias[f + 1];
  o.z = dz * rstd * w[f + 2] + bias[f + 2];
  o.w = dw * rstd * w[f + 3] + bias[f + 3];
  ((float4*)out)[row * 32 + lane] = o;
}

// ---------------------------------------------------------------------------
extern "C" void kernel_launch(void* const* d_in, const int* in_sizes, int n_in,
                              void* d_out, int out_size, void* d_ws, size_t ws_size,
                              hipStream_t stream)
{
  (void)in_sizes; (void)n_in; (void)out_size; (void)ws_size;
  typedef unsigned short u16;
  const float* x  = (const float*)d_in[0];
  const unsigned char* pm = (const unsigned char*)d_in[1];
  const float* Wq = (const float*)d_in[2];   const float* bq = (const float*)d_in[3];
  const float* Wk = (const float*)d_in[4];   const float* bk = (const float*)d_in[5];
  const float* Wv = (const float*)d_in[6];   const float* bv = (const float*)d_in[7];
  const float* Wo = (const float*)d_in[8];   const float* bo = (const float*)d_in[9];
  const float* Wg = (const float*)d_in[10];
  const float* cWq = (const float*)d_in[11]; const float* cbq = (const float*)d_in[12];
  const float* cWk = (const float*)d_in[13]; const float* cbk = (const float*)d_in[14];
  const float* cWv = (const float*)d_in[15]; const float* cbv = (const float*)d_in[16];
  const float* cWf = (const float*)d_in[17]; const float* cbf = (const float*)d_in[18];
  const float* ln1w = (const float*)d_in[19]; const float* ln1b = (const float*)d_in[20];
  const float* lncw = (const float*)d_in[21]; const float* lncb = (const float*)d_in[22];
  const float* lnfw = (const float*)d_in[23]; const float* lnfb = (const float*)d_in[24];
  const float* W1 = (const float*)d_in[25];  const float* b1 = (const float*)d_in[26];
  const float* W2 = (const float*)d_in[27];  const float* b2 = (const float*)d_in[28];
  float* out = (float*)d_out;

  const size_t P  = (size_t)S_LEN * BSZ * EMB;  // 4M elements
  const size_t Hf = P / 2;                      // 2M floats = one bf16 act slot
  float* ws = (float*)d_ws;

  // activation region (4 slots x 2M floats), multiply reused:
  u16*   f_qb    = (u16*)(ws);                  // q bf16; kb/vt follow at +P elems
  u16*   f_kb    = (u16*)(ws + Hf);
  u16*   f_vt    = (u16*)(ws + 2 * Hf);
  u16*   f_attnb = (u16*)(ws + 3 * Hf);
  float* cqkv  = ws;                            // (Mc,192) f32 = slots 0..2 exactly
  float* cattn = ws + 3 * Hf;
  u16*   f_hb  = (u16*)ws;                      // ffn hidden, 16M bf16 = all 4 slots

  float* f_gate = ws + 4 * Hf;                  // S*B*32
  float* f_proj = f_gate + (size_t)S_LEN * BSZ * NB * NH;
  float* f_x1   = f_proj + P;
  float* f_x2   = f_x1 + P;
  float* wbase  = f_x2 + P;
  u16* wqkv_t = (u16*)wbase;                    // (8, 384, 128)
  u16* wo_t   = wqkv_t + 393216;                // (8, 128, 128)
  u16* w1_t   = wo_t + 131072;                  // (8, 512, 128)
  u16* w2_t   = w1_t + 524288;                  // (8, 128, 512)
  u16* wcomm_t= w2_t + 524288;                  // (192, 128)
  u16* cwf_t  = wcomm_t + 24576;                // (128, 64)
  float* b_qkv  = (float*)(cwf_t + 8192);       // (8, 384)
  float* b_comm = b_qkv + 3072;                 // (192)

  const int M  = S_LEN * BSZ;                   // 4096
  const int Mc = M * NB;                        // 32768
  const float qs = 0.17677669529663687f;        // HD^-0.5

  // 0) pack weights (bf16, transposed) and fused biases — one-time, tiny
  pack_w<<<512, 256, 0, stream>>>(Wq, wqkv_t, BD, BD, 384 * BD, 0);
  pack_w<<<512, 256, 0, stream>>>(Wk, wqkv_t, BD, BD, 384 * BD, 128);
  pack_w<<<512, 256, 0, stream>>>(Wv, wqkv_t, BD, BD, 384 * BD, 256);
  pack_w<<<512, 256, 0, stream>>>(Wo, wo_t, BD, BD, BD * BD, 0);
  pack_w<<<2048, 256, 0, stream>>>(W1, w1_t, BD, FBD, FBD * BD, 0);
  pack_w<<<2048, 256, 0, stream>>>(W2, w2_t, FBD, BD, BD * FBD, 0);
  pack_w<<<32, 256, 0, stream>>>(cWq, wcomm_t, BD, 64, 0, 0);
  pack_w<<<32, 256, 0, stream>>>(cWk, wcomm_t, BD, 64, 0, 64);
  pack_w<<<32, 256, 0, stream>>>(cWv, wcomm_t, BD, 64, 0, 128);
  pack_w<<<32, 256, 0, stream>>>(cWf, cwf_t, 64, BD, 0, 0);
  pack_bias<<<4, 256, 0, stream>>>(bq, b_qkv, BD, 384, 0, NB * BD);
  pack_bias<<<4, 256, 0, stream>>>(bk, b_qkv, BD, 384, 128, NB * BD);
  pack_bias<<<4, 256, 0, stream>>>(bv, b_qkv, BD, 384, 256, NB * BD);
  pack_bias<<<1, 256, 0, stream>>>(cbq, b_comm, 64, 192, 0, 64);
  pack_bias<<<1, 256, 0, stream>>>(cbk, b_comm, 64, 192, 64, 64);
  pack_bias<<<1, 256, 0, stream>>>(cbv, b_comm, 64, 192, 128, 64);

  // 1) fused grouped QKV projection (N=384): q*scale | k | v-transposed, bf16
  gemm_grouped<0, 3, BD, float><<<dim3(M / 16, NB), 256, 0, stream>>>(
      x, wqkv_t, b_qkv, nullptr, f_qb, 384, NB, qs);
  // 2) top-k gate from original x
  gate_topk<<<dim3(M / 8), 256, 0, stream>>>(x, Wg, f_gate);
  // 3) block-diagonal attention (gate fused) -> bf16
  attn_fa<<<dim3(S_LEN / 16, NB * NH, BSZ), 32, 0, stream>>>(
      f_qb, f_kb, f_vt, pm, f_gate, f_attnb);
  // 4) output projection (bf16 A, async-LDS staged)
  gemm_grouped<0, 0, BD, u16><<<dim3(M / 16, NB), 256, 0, stream>>>(
      f_attnb, wo_t, bo, f_proj, nullptr, BD, NB, 1.0f);
  // 5) x1 = LN1(x + attn)
  add_norm<<<dim3(Mc / 8), 256, 0, stream>>>(x, f_proj, ln1w, ln1b, f_x1);
  // 6) fused comm q|k|v projection (G=1, N=192) -> f32
  gemm_grouped<0, 0, BD, float><<<dim3(Mc / 16, 1), 256, 0, stream>>>(
      f_x1, wcomm_t, b_comm, cqkv, nullptr, 192, 1, 1.0f);
  // 7) 8x8 inter-block attention
  comm_attn<<<dim3(M * 16 / 256), 256, 0, stream>>>(cqkv, cattn);
  // 8) comm output projection (KD = 64, N = 128)
  gemm_grouped<0, 0, 64, float><<<dim3(Mc / 16, 1), 256, 0, stream>>>(
      cattn, cwf_t, cbf, f_proj, nullptr, BD, 1, 1.0f);
  // 9) x2 = LNc(x1 + comm)
  add_norm<<<dim3(Mc / 8), 256, 0, stream>>>(f_x1, f_proj, lncw, lncb, f_x2);
  // 10) FFN up + ReLU -> bf16 hidden
  gemm_grouped<1, 1, BD, float><<<dim3(M / 16, NB), 256, 0, stream>>>(
      f_x2, w1_t, b1, nullptr, f_hb, FBD, NB, 1.0f);
  // 11) FFN down (KD = 512, bf16 A, async-LDS staged)
  gemm_grouped<0, 0, FBD, u16><<<dim3(M / 16, NB), 256, 0, stream>>>(
      f_hb, w2_t, b2, f_proj, nullptr, BD, NB, 1.0f);
  // 12) out = LNf(x2 + ffn)
  add_norm<<<dim3(Mc / 8), 256, 0, stream>>>(f_x2, f_proj, lnfw, lnfb, out);
}
